// FindArgModel_43971875177071
// MI455X (gfx1250) — matrix operand order repaired and checked by reading, not compiled, not run on hardware
//
#include <hip/hip_runtime.h>
#include <hip/hip_bf16.h>

// GRU forward (B=128, S=512, H=512, V=32000) on gfx1250 (MI455X).
// Pipeline:
//   prep:   f32->bf16 weight conversion, h0=0, lik=b_l, barrier ctr=0
//   gx:     gx[t][b][3H] = relu(emb[tok]) @ W_ih^T + b_ih   (bf16 WMMA, whole device)
//   recur:  513 sequential GRU steps, persistent 32-block grid, 1 grid-sync/step,
//           3 fused gate GEMMs per wave-tile (v_wmma_f32_16x16x32_bf16),
//           fused relu(h)@W_l likelihood head via shfl reductions + f32 atomics
//   lsm:    reorder [EOS, 0..S-1] and log_softmax over S+1 per batch row

typedef __attribute__((ext_vector_type(16))) __bf16 v16bf;
typedef __attribute__((ext_vector_type(8)))  __bf16 v8bf;
typedef __attribute__((ext_vector_type(8)))  float  v8f;
typedef __attribute__((ext_vector_type(4)))  float  v4f;

#define Bsz  128
#define Ssz  512
#define Hsz  512
#define G3H  1536
#define TT   513      // S+1 timesteps (t==512 is the EOS step)
#define EOSTOK 1
#define RBLK 32       // persistent blocks in the recurrence kernel

static __device__ __forceinline__ float sigmoidf_(float x) {
    return 1.0f / (1.0f + __expf(-x));
}

// ---------------------------------------------------------------- prep ----
__global__ void prep_kernel(const float* __restrict__ Wih, const float* __restrict__ Whh,
                            const float* __restrict__ bl,
                            __bf16* __restrict__ WihB, __bf16* __restrict__ WhhB,
                            float* __restrict__ hf, __bf16* __restrict__ hb,
                            float* __restrict__ lik, unsigned* __restrict__ ctr)
{
    size_t i0 = (size_t)blockIdx.x * blockDim.x + threadIdx.x;
    size_t stride = (size_t)gridDim.x * blockDim.x;
    for (size_t i = i0; i < (size_t)G3H * Hsz; i += stride) {
        WihB[i] = (__bf16)Wih[i];
        WhhB[i] = (__bf16)Whh[i];
    }
    for (size_t i = i0; i < (size_t)2 * Bsz * Hsz; i += stride) {  // both h buffers
        hf[i] = 0.0f;
        hb[i] = (__bf16)0.0f;
    }
    float blv = bl[0];
    for (size_t i = i0; i < (size_t)TT * Bsz; i += stride) lik[i] = blv;
    if (i0 == 0) *ctr = 0u;
}

// ------------------------------------------------------------------ gx ----
// Grid: 513 * 8(btiles) * 12(jgroups) blocks, 256 threads (8 waves).
// Each wave: one 16x16 f32 tile of gx[t], K=512 in 16 bf16 WMMA steps.
// Waves in a block share (t, btile) -> embedding gather rows hit L0.
__global__ void gx_kernel(const int* __restrict__ goal,
                          const float* __restrict__ emb,
                          const __bf16* __restrict__ WihB,
                          const float* __restrict__ bih,
                          __bf16* __restrict__ gx)
{
    const int lane = threadIdx.x & 31;
    const int wave = threadIdx.x >> 5;
    const int bi    = blockIdx.x;
    const int t     = bi / 96;
    const int btile = (bi % 96) / 12;
    const int jgrp  = bi % 12;
    const int jtile = jgrp * 8 + wave;

    const int n  = lane & 15;
    const int hi = lane >> 4;
    const int b  = btile * 16 + n;     // A row handled by this lane
    const int j  = jtile * 16 + n;     // B column (= W_ih row) for this lane

    const int tok = (t < Ssz) ? goal[(size_t)b * Ssz + t] : EOSTOK;
    const float*  erow = emb  + (size_t)tok * Hsz;
    const __bf16* wrow = WihB + (size_t)j   * Hsz;

    v8f acc = {};
    for (int k = 0; k < Hsz; k += 32) {
        // A operand in contiguous k order (matches B): lane half hi owns 16 contiguous K [kb..kb+15]
        const int kb = k + hi * 16;
        v4f e0 = *(const v4f*)(erow + kb);
        v4f e1 = *(const v4f*)(erow + kb + 4);
        v4f e2 = *(const v4f*)(erow + kb + 8);
        v4f e3 = *(const v4f*)(erow + kb + 12);
        v16bf a;
        #pragma unroll
        for (int i = 0; i < 4; ++i) {
            a[i]      = (__bf16)fmaxf(e0[i], 0.0f);
            a[4 + i]  = (__bf16)fmaxf(e1[i], 0.0f);
            a[8 + i]  = (__bf16)fmaxf(e2[i], 0.0f);
            a[12 + i] = (__bf16)fmaxf(e3[i], 0.0f);
        }
        // B operand in the same contiguous k order as A: lane half hi owns 16 contiguous K, one 32B load
        v16bf bm = *(const v16bf*)(wrow + k + hi * 16);
        acc = __builtin_amdgcn_wmma_f32_16x16x32_bf16(false, a, false, bm,
                                                      (short)0, acc, false, false);
    }
    const float bj = bih[j];
    #pragma unroll
    for (int v = 0; v < 8; ++v) {
        const int brow = btile * 16 + v + hi * 8;   // C-layout: VGPR v = row v / v+8
        gx[((size_t)t * Bsz + brow) * G3H + j] = (__bf16)(acc[v] + bj);
    }
}

// --------------------------------------------------------------- recur ----
__global__ void recur_kernel(const __bf16* __restrict__ WhhB,
                             const float* __restrict__ bhh,
                             const float* __restrict__ Wl,
                             const __bf16* __restrict__ gx,
                             float* __restrict__ hf,    // 2 x [128][512] f32 (double buffer)
                             __bf16* __restrict__ hb,   // 2 x [128][512] bf16
                             float* __restrict__ lik,   // [513][128], pre-init to b_l
                             unsigned* __restrict__ ctr)
{
    const int lane  = threadIdx.x & 31;
    const int w     = blockIdx.x * 8 + (threadIdx.x >> 5);  // 0..255 waves
    const int btile = w >> 5;                               // 0..7
    const int jtile = w & 31;                               // 0..31
    const int n     = lane & 15;
    const int hi    = lane >> 4;
    const int j     = jtile * 16 + n;
    const int b0    = btile * 16;

    const __bf16* wr = WhhB + (size_t)j            * Hsz;
    const __bf16* wz = WhhB + (size_t)(j + Hsz)    * Hsz;
    const __bf16* wn = WhhB + (size_t)(j + 2*Hsz)  * Hsz;
    const float bhr = bhh[j], bhz = bhh[j + Hsz], bhn = bhh[j + 2*Hsz];
    const float wl  = Wl[j];

    for (int t = 0; t < TT; ++t) {
        const __bf16* hin   = hb + (size_t)(t & 1)       * (Bsz * Hsz);
        const float*  hfin  = hf + (size_t)(t & 1)       * (Bsz * Hsz);
        __bf16*       hout  = hb + (size_t)((t + 1) & 1) * (Bsz * Hsz);
        float*        hfout = hf + (size_t)((t + 1) & 1) * (Bsz * Hsz);

        v8f ar = {}, az = {}, an = {};
        const __bf16* hrow = hin + (size_t)(b0 + n) * Hsz;   // A row for this lane
        for (int k = 0; k < Hsz; k += 32) {
            const int kb = k + hi * 16;
            union { v16bf v; v8bf h[2]; } a;
            a.h[0] = *(const v8bf*)(hrow + kb);
            a.h[1] = *(const v8bf*)(hrow + kb + 8);
            const int kw = k + hi * 16;
            v16bf br = *(const v16bf*)(wr + kw);
            v16bf bz = *(const v16bf*)(wz + kw);
            v16bf bn = *(const v16bf*)(wn + kw);
            ar = __builtin_amdgcn_wmma_f32_16x16x32_bf16(false, a.v, false, br, (short)0, ar, false, false);
            az = __builtin_amdgcn_wmma_f32_16x16x32_bf16(false, a.v, false, bz, (short)0, az, false, false);
            an = __builtin_amdgcn_wmma_f32_16x16x32_bf16(false, a.v, false, bn, (short)0, an, false, false);
        }

        const __bf16* gxt = gx + ((size_t)t * Bsz) * G3H;
        #pragma unroll
        for (int v = 0; v < 8; ++v) {
            const int brow = b0 + v + hi * 8;
            const size_t gb = (size_t)brow * G3H + j;
            const float xr = (float)gxt[gb];
            const float xz = (float)gxt[gb + Hsz];
            const float xn = (float)gxt[gb + 2*Hsz];
            const float r  = sigmoidf_(xr + ar[v] + bhr);
            const float z  = sigmoidf_(xz + az[v] + bhz);
            const float nn = tanhf(xn + r * (an[v] + bhn));
            const float hold = hfin[(size_t)brow * Hsz + j];
            const float hnew = (1.0f - z) * nn + z * hold;
            hfout[(size_t)brow * Hsz + j] = hnew;
            hout [(size_t)brow * Hsz + j] = (__bf16)hnew;

            // likelihood head: sum_j relu(h)*W_l  -> butterfly over the 16 columns
            float p = fmaxf(hnew, 0.0f) * wl;
            p += __shfl_xor(p, 1, 16);
            p += __shfl_xor(p, 2, 16);
            p += __shfl_xor(p, 4, 16);
            p += __shfl_xor(p, 8, 16);
            if (n == 0) atomicAdd(lik + (size_t)t * Bsz + brow, p);
        }

        // ---- one grid-wide barrier per step (monotonic counter) ----
        __threadfence();
        __syncthreads();
        if (threadIdx.x == 0) {
            __hip_atomic_fetch_add(ctr, 1u, __ATOMIC_RELEASE, __HIP_MEMORY_SCOPE_AGENT);
            const unsigned target = (unsigned)RBLK * (unsigned)(t + 1);
            while (__hip_atomic_load(ctr, __ATOMIC_ACQUIRE, __HIP_MEMORY_SCOPE_AGENT) < target)
                __builtin_amdgcn_s_sleep(2);
        }
        __syncthreads();
    }
}

// ----------------------------------------------------------------- lsm ----
__global__ void lsm_kernel(const float* __restrict__ lik, float* __restrict__ out)
{
    __shared__ float red[256];
    const int b   = blockIdx.x;
    const int tid = threadIdx.x;

    float m = -3.4e38f;
    for (int s = tid; s < TT; s += 256) {
        const int t = (s == 0) ? Ssz : (s - 1);
        m = fmaxf(m, lik[(size_t)t * Bsz + b]);
    }
    red[tid] = m; __syncthreads();
    for (int o = 128; o > 0; o >>= 1) {
        if (tid < o) red[tid] = fmaxf(red[tid], red[tid + o]);
        __syncthreads();
    }
    m = red[0]; __syncthreads();

    float acc = 0.0f;
    for (int s = tid; s < TT; s += 256) {
        const int t = (s == 0) ? Ssz : (s - 1);
        acc += __expf(lik[(size_t)t * Bsz + b] - m);
    }
    red[tid] = acc; __syncthreads();
    for (int o = 128; o > 0; o >>= 1) {
        if (tid < o) red[tid] += red[tid + o];
        __syncthreads();
    }
    const float lse = m + __logf(red[0]);

    for (int s = tid; s < TT; s += 256) {
        const int t = (s == 0) ? Ssz : (s - 1);
        out[(size_t)b * TT + s] = lik[(size_t)t * Bsz + b] - lse;
    }
}

// -------------------------------------------------------------- launch ----
extern "C" void kernel_launch(void* const* d_in, const int* in_sizes, int n_in,
                              void* d_out, int out_size, void* d_ws, size_t ws_size,
                              hipStream_t stream)
{
    (void)in_sizes; (void)n_in; (void)out_size; (void)ws_size;
    const int*   goal = (const int*)  d_in[0];
    const float* emb  = (const float*)d_in[1];
    const float* Wih  = (const float*)d_in[2];
    const float* Whh  = (const float*)d_in[3];
    const float* bih  = (const float*)d_in[4];
    const float* bhh  = (const float*)d_in[5];
    const float* Wl   = (const float*)d_in[6];
    const float* bl   = (const float*)d_in[7];
    float* out = (float*)d_out;

    char* ws = (char*)d_ws;
    size_t off = 0;
    auto alloc = [&](size_t bytes) -> char* {
        off = (off + 255) & ~(size_t)255;
        char* p = ws + off;
        off += bytes;
        return p;
    };
    __bf16*   WihB = (__bf16*)  alloc((size_t)G3H * Hsz * 2);
    __bf16*   WhhB = (__bf16*)  alloc((size_t)G3H * Hsz * 2);
    float*    hf   = (float*)   alloc((size_t)2 * Bsz * Hsz * 4);
    __bf16*   hb   = (__bf16*)  alloc((size_t)2 * Bsz * Hsz * 2);
    float*    lik  = (float*)   alloc((size_t)TT * Bsz * 4);
    unsigned* ctr  = (unsigned*)alloc(256);
    __bf16*   gx   = (__bf16*)  alloc((size_t)TT * Bsz * G3H * 2);  // ~202 MB

    hipLaunchKernelGGL(prep_kernel,  dim3(2048),    dim3(256), 0, stream,
                       Wih, Whh, bl, WihB, WhhB, hf, hb, lik, ctr);
    hipLaunchKernelGGL(gx_kernel,    dim3(TT * 96), dim3(256), 0, stream,
                       goal, emb, WihB, bih, gx);
    hipLaunchKernelGGL(recur_kernel, dim3(RBLK),    dim3(256), 0, stream,
                       WhhB, bhh, Wl, gx, hf, hb, lik, ctr);
    hipLaunchKernelGGL(lsm_kernel,   dim3(Bsz),     dim3(256), 0, stream,
                       lik, out);
}
